// NTXent_43327630082371
// MI455X (gfx1250) — matrix operand order repaired and checked by reading, not compiled, not run on hardware
//
#include <hip/hip_runtime.h>
#include <hip/hip_bf16.h>

// ---------------------------------------------------------------------------
// NT-Xent loss, fused flash-softmax GEMM for MI455X (gfx1250, wave32, WMMA).
// z1,z2: [4096,256] f32.  zn(f16) = row-normalized [8192,256] (4MB, L2-resident).
// sim = zn @ zn^T * 2 is never materialized.  Since |cos| <= 1, logits are
// bounded by 2.0 => fixed-max logsumexp: lse = 2 + log(sum exp(x-2)).
// loss = mean_i( lse_i - sim[i, i^4096] ).
// B tiles are double-buffered (ping-pong, 2x64 VGPRs) so the 16 b128 loads of
// tile jt+1 overlap the 8 WMMAs of tile jt.  Tile-type tests go through
// readfirstlane so they compile to scalar branches (no exec-mask churn).
// ---------------------------------------------------------------------------

typedef __attribute__((ext_vector_type(16))) _Float16 v16h;
typedef __attribute__((ext_vector_type(8)))  _Float16 v8h;
typedef __attribute__((ext_vector_type(4)))  _Float16 v4h;
typedef __attribute__((ext_vector_type(8)))  float    v8f;

#define NTX_N    4096
#define NTX_2N   8192
#define NTX_D    256
#define NTX_EPS  1e-8f
#define NTX_INVT 2.0f          // 1 / temperature(0.5); also the logit bound
#define NTX_L2E  1.44269504088896340736f

// exp2 on the v_exp_f32 transcendental unit
#define NTX_EXP2(x) __builtin_amdgcn_exp2f(x)

// ---------------------------------------------------------------------------
// Kernel 1: row-normalize, f32 -> f16.  One wave per row.
// ---------------------------------------------------------------------------
__global__ __launch_bounds__(256) void ntxent_normalize(
    const float* __restrict__ z1, const float* __restrict__ z2,
    _Float16* __restrict__ zn)
{
    const int wave = (blockIdx.x * 256 + threadIdx.x) >> 5;   // row 0..8191
    const int lane = threadIdx.x & 31;
    if (wave >= NTX_2N) return;

    const float* src = (wave < NTX_N) ? (z1 + (size_t)wave * NTX_D)
                                      : (z2 + (size_t)(wave - NTX_N) * NTX_D);
    float4 a = ((const float4*)src)[lane];
    float4 b = ((const float4*)src)[lane + 32];
    float ss = a.x*a.x + a.y*a.y + a.z*a.z + a.w*a.w
             + b.x*b.x + b.y*b.y + b.z*b.z + b.w*b.w;
    #pragma unroll
    for (int off = 16; off > 0; off >>= 1) ss += __shfl_xor(ss, off, 32);

    const float scale = 1.0f / fmaxf(sqrtf(ss), NTX_EPS);
    v4h ha = { (_Float16)(a.x*scale), (_Float16)(a.y*scale),
               (_Float16)(a.z*scale), (_Float16)(a.w*scale) };
    v4h hb = { (_Float16)(b.x*scale), (_Float16)(b.y*scale),
               (_Float16)(b.z*scale), (_Float16)(b.w*scale) };
    v4h* dst = (v4h*)(zn + (size_t)wave * NTX_D);
    dst[lane]      = ha;
    dst[lane + 32] = hb;
}

// ---------------------------------------------------------------------------
// Kernel 2: fused sim GEMM + fixed-max logsumexp.  One wave per 16-row tile.
// A fragments (16 rows x 256 K, f16) stay resident in 64 VGPRs.
// ---------------------------------------------------------------------------
__global__ __launch_bounds__(256) void ntxent_sim_lse(
    const _Float16* __restrict__ zn,
    float* __restrict__ lse_out,   // [8192]
    float* __restrict__ pair_out)  // [8192]
{
    const int gwave = (blockIdx.x * 256 + threadIdx.x) >> 5;  // row tile 0..511
    const int lane  = threadIdx.x & 31;
    const int half  = lane >> 4;
    const int laneN = lane & 15;

    const int rt = gwave;               // this wave's row tile (diag tile)
    const int r0 = rt << 4;
    // wave-uniform copies for scalar (s_cmp) branching in the sweep loop
    const int rt_u = __builtin_amdgcn_readfirstlane(rt);
    const int pt_u = rt_u ^ 256;        // pair tile: (i ^ 4096) >> 4

    // --- resident A fragments: rows r0..r0+15, K = 0..255 -------------------
    // 16-bit A 16x32 layout: lanes 0-15 hold K=[0..7]+[16..23] of row laneN,
    // lanes 16-31 hold K=[8..15]+[24..31].
    v16h afrag[8];
    {
        const _Float16* arow = zn + (size_t)(r0 + laneN) * NTX_D + half * 8;
        #pragma unroll
        for (int kk = 0; kk < 8; ++kk) {
            v8h lo = *(const v8h*)(arow + kk * 32);
            v8h hi = *(const v8h*)(arow + kk * 32 + 16);
            afrag[kk] = __builtin_shufflevector(lo, hi,
                0,1,2,3,4,5,6,7,8,9,10,11,12,13,14,15);
        }
    }

    // B tile address: lane holds column n = laneN; lanes 0-15 carry K=[0..7]+[16..23],
    // lanes 16-31 carry K=[8..15]+[24..31] (same K order as A).
    const _Float16* bbase = zn + (size_t)laneN * NTX_D + half * 8;

    float s[8];
    #pragma unroll
    for (int v = 0; v < 8; ++v) s[v] = 0.0f;

    v16h b0[8], b1[8];

    // load one column tile (16 b128 loads, single s_clause)
    auto load_btile = [&](int jt, v16h (&bf)[8]) {
        const _Float16* p = bbase + (size_t)(jt << 4) * NTX_D;
        #pragma unroll
        for (int kk = 0; kk < 8; ++kk) {
            v8h lo = *(const v8h*)(p + kk * 32);
            v8h hi = *(const v8h*)(p + kk * 32 + 16);
            bf[kk] = __builtin_shufflevector(lo, hi,
                0,1,2,3,4,5,6,7,8,9,10,11,12,13,14,15);
        }
    };

    // 8 WMMAs (two independent accumulator chains) + bounded-max softmax.
    // exp(2c - 2) = exp2(fma(c, 2*log2e, -2*log2e)): one fma + one v_exp_f32.
    auto process = [&](int jt, const v16h (&bf)[8]) {
        v8f c0 = {0.f,0.f,0.f,0.f,0.f,0.f,0.f,0.f};
        v8f c1 = {0.f,0.f,0.f,0.f,0.f,0.f,0.f,0.f};
        #pragma unroll
        for (int kk = 0; kk < 8; kk += 2) {
            c0 = __builtin_amdgcn_wmma_f32_16x16x32_f16(
                     false, afrag[kk],   false, bf[kk],   (short)0, c0, false, false);
            c1 = __builtin_amdgcn_wmma_f32_16x16x32_f16(
                     false, afrag[kk+1], false, bf[kk+1], (short)0, c1, false, false);
        }
        // C layout: VGPR v -> row (v + 8*half), col = laneN.
        if (jt == rt_u) {                     // diagonal tile: mask col==row
            #pragma unroll
            for (int v = 0; v < 8; ++v) {
                float x = c0[v] + c1[v];
                if (laneN == v + (half << 3)) x = -1.0e30f;   // exp2 -> 0
                s[v] += NTX_EXP2(fmaf(x, NTX_INVT * NTX_L2E, -NTX_INVT * NTX_L2E));
            }
        } else if (jt == pt_u) {              // pair tile: capture positives
            #pragma unroll
            for (int v = 0; v < 8; ++v) {
                const int rowl = v + (half << 3);
                float x = c0[v] + c1[v];
                if (laneN == rowl) pair_out[r0 + rowl] = x * NTX_INVT;
                s[v] += NTX_EXP2(fmaf(x, NTX_INVT * NTX_L2E, -NTX_INVT * NTX_L2E));
            }
        } else {                              // common path (510/512 tiles)
            #pragma unroll
            for (int v = 0; v < 8; ++v)
                s[v] += NTX_EXP2(fmaf(c0[v] + c1[v],
                                      NTX_INVT * NTX_L2E, -NTX_INVT * NTX_L2E));
        }
    };

    // --- ping-pong sweep over all 512 column tiles --------------------------
    load_btile(0, b0);
    for (int jt = 0; jt < 512; jt += 2) {
        load_btile(jt + 1, b1);               // in flight during process(jt)
        process(jt, b0);
        if (jt + 2 < 512) load_btile(jt + 2, b0);   // in flight during jt+1
        process(jt + 1, b1);
    }

    // --- sum s across the 16 lanes sharing a row ----------------------------
    #pragma unroll
    for (int v = 0; v < 8; ++v) {
        float sv = s[v];
        #pragma unroll
        for (int off = 1; off < 16; off <<= 1) sv += __shfl_xor(sv, off, 32);
        if (laneN == 0)
            lse_out[r0 + v + (half << 3)] = NTX_INVT + __logf(sv);
    }
}

// ---------------------------------------------------------------------------
// Kernel 3: loss = mean_i(lse_i - pair_i).  One block.
// ---------------------------------------------------------------------------
__global__ __launch_bounds__(256) void ntxent_reduce(
    const float* __restrict__ lse, const float* __restrict__ pair,
    float* __restrict__ out)
{
    __shared__ float red[8];
    float acc = 0.0f;
    for (int i = threadIdx.x; i < NTX_2N; i += 256)
        acc += lse[i] - pair[i];
    #pragma unroll
    for (int off = 16; off > 0; off >>= 1) acc += __shfl_xor(acc, off, 32);
    if ((threadIdx.x & 31) == 0) red[threadIdx.x >> 5] = acc;
    __syncthreads();
    if (threadIdx.x == 0) {
        float t = 0.0f;
        #pragma unroll
        for (int i = 0; i < 8; ++i) t += red[i];
        out[0] = t * (1.0f / (float)NTX_2N);
    }
}

// ---------------------------------------------------------------------------
extern "C" void kernel_launch(void* const* d_in, const int* in_sizes, int n_in,
                              void* d_out, int out_size, void* d_ws, size_t ws_size,
                              hipStream_t stream)
{
    const float* z1 = (const float*)d_in[0];
    const float* z2 = (const float*)d_in[1];
    float* out = (float*)d_out;

    // workspace carve-out
    _Float16* zn   = (_Float16*)d_ws;                               // 4 MB
    float*    lse  = (float*)((char*)d_ws + (size_t)NTX_2N * NTX_D * sizeof(_Float16));
    float*    pair = lse + NTX_2N;

    ntxent_normalize<<<NTX_2N / 8, 256, 0, stream>>>(z1, z2, zn);
    ntxent_sim_lse<<<512 / 8, 256, 0, stream>>>(zn, lse, pair);
    ntxent_reduce<<<1, 256, 0, stream>>>(lse, pair, out);
}
